// GraphResBlock_62843961475245
// MI455X (gfx1250) — matrix operand order repaired and checked by reading, not compiled, not run on hardware
//
#include <hip/hip_runtime.h>
#include <hip/hip_bf16.h>

typedef unsigned short ushort_t;
typedef unsigned int uint_t;

typedef __bf16 v16bf __attribute__((ext_vector_type(16)));
typedef float  v8f   __attribute__((ext_vector_type(8)));
typedef int    v4i_t __attribute__((ext_vector_type(4)));

#define N_NODES 100000
#define NPAD    100032      // N rounded up to 64 (GEMM tile rows)
#define N_EDGES 700000
#define C 256
#define EMB_CH 512
#define NET 7
#define NNT 7
#define BATCH 8
#define GROUP 32
#define CPG 8
#define EPSV 1e-5f
#define CP 263
#define CPAD 264
#define KP 1848             // 7*264 (real K)
#define KPP 1856            // padded K (58 * 32), also S row stride
#define NKT 58

#if defined(__has_builtin)
#if __has_builtin(__builtin_amdgcn_global_load_async_to_lds_b128) && \
    __has_builtin(__builtin_amdgcn_s_wait_asynccnt)
#define ASYNC_OK 1
#else
#define ASYNC_OK 0
#endif
#else
#define ASYNC_OK 0
#endif

// param1: v4i in global (__device__) AS ; param2: LDS AS3
#define AS1V(p) ((__attribute__((address_space(1))) v4i_t*)(p))
#define AS3V(p) ((__attribute__((address_space(3))) v4i_t*)(p))

__device__ __forceinline__ float silu(float v) { return v / (1.f + __expf(-v)); }

struct alignas(16) U4 { uint_t x, y, z, w; };
union FragBF { U4 u[2]; v16bf v; };

// ---------- per-batch node counts ----------
__global__ void k_batch_count(const int* __restrict__ batch_id, int* __restrict__ bcnt) {
    __shared__ int lc[BATCH];
    if (threadIdx.x < BATCH) lc[threadIdx.x] = 0;
    __syncthreads();
    for (int n = blockIdx.x * 256 + threadIdx.x; n < N_NODES; n += gridDim.x * 256)
        atomicAdd(&lc[batch_id[n]], 1);
    __syncthreads();
    if (threadIdx.x < BATCH) atomicAdd(&bcnt[threadIdx.x], lc[threadIdx.x]);
}

// ---------- per (batch,group) sum & sumsq over [N,256] ----------
__global__ void k_stats(const float* __restrict__ data, const int* __restrict__ batch_id,
                        float* __restrict__ ssum, float* __restrict__ ssq) {
    __shared__ float ls[BATCH * GROUP];
    __shared__ float lq[BATCH * GROUP];
    const int tid = threadIdx.x;
    ls[tid] = 0.f; lq[tid] = 0.f;
    __syncthreads();
    const int g = tid >> 3;
    const int nbeg = blockIdx.x * 128;
    int nend = nbeg + 128; if (nend > N_NODES) nend = N_NODES;
    float s = 0.f, q = 0.f; int cb = -1;
    for (int n = nbeg; n < nend; ++n) {
        int b = batch_id[n];
        if (b != cb) {
            if (cb >= 0) { atomicAdd(&ls[cb * GROUP + g], s); atomicAdd(&lq[cb * GROUP + g], q); }
            s = 0.f; q = 0.f; cb = b;
        }
        float v = data[(size_t)n * C + tid];
        s += v; q += v * v;
    }
    if (cb >= 0) { atomicAdd(&ls[cb * GROUP + g], s); atomicAdd(&lq[cb * GROUP + g], q); }
    __syncthreads();
    atomicAdd(&ssum[tid], ls[tid]);
    atomicAdd(&ssq[tid], lq[tid]);
}

// ---------- mean & inv_std per (batch,group) ----------
__global__ void k_finalize(const float* __restrict__ ssum, const float* __restrict__ ssq,
                           const int* __restrict__ bcnt,
                           float* __restrict__ mean, float* __restrict__ istd) {
    int tid = threadIdx.x;
    int b = tid >> 5;
    float D  = (float)bcnt[b] * (float)CPG;
    float ic = 1.f / (D + EPSV);
    float S  = ssum[tid];
    float m  = S * ic;
    float var = (ssq[tid] - 2.f * m * S + D * m * m) * ic;
    mean[tid] = m;
    istd[tid] = rsqrtf(var + EPSV);
}

// ---------- normalize + silu + one-hot concat -> [N,264] ----------
__global__ void k_norm_act(const float* __restrict__ in, const int* __restrict__ batch_id,
                           const int* __restrict__ node_type,
                           const float* __restrict__ mean, const float* __restrict__ istd,
                           const float* __restrict__ gw, const float* __restrict__ gb,
                           float* __restrict__ outp) {
    const int n = blockIdx.x, c = threadIdx.x;
    const int b = batch_id[n], g = c >> 3;
    float v = in[(size_t)n * C + c];
    v = (v - mean[b * GROUP + g]) * istd[b * GROUP + g] * gw[c] + gb[c];
    outp[(size_t)n * CPAD + c] = silu(v);
    if (c < 8) outp[(size_t)n * CPAD + C + c] = (c < NNT && node_type[n] == c) ? 1.f : 0.f;
}

// ---------- edge segment counts & inverse ----------
__global__ void k_count_edges(const int* __restrict__ row, const int* __restrict__ et,
                              int* __restrict__ cnt) {
    int e = blockIdx.x * 256 + threadIdx.x;
    if (e < N_EDGES) atomicAdd(&cnt[row[e] * NET + et[e]], 1);
}
__global__ void k_invcnt(const int* __restrict__ cnt, float* __restrict__ ic) {
    int i = blockIdx.x * 256 + threadIdx.x;
    if (i < N_NODES * NET) { int c = cnt[i]; ic[i] = 1.f / (float)(c > 0 ? c : 1); }
}

// ---------- scatter-add gathered rows, pre-scaled by 1/count ----------
__global__ void k_scatter(const float* __restrict__ h, const int* __restrict__ row,
                          const int* __restrict__ col, const int* __restrict__ et,
                          const float* __restrict__ invc, float* __restrict__ S) {
    int wid = (blockIdx.x * blockDim.x + threadIdx.x) >> 5;
    int lane = threadIdx.x & 31;
    if (wid >= N_EDGES) return;
    int r = row[wid];
    int t = et[wid];
    float scale = invc[r * NET + t];
    const float* src = h + (size_t)col[wid] * CPAD;
    float* dst = S + (size_t)r * KPP + t * CPAD;
    for (int c = lane; c < CP; c += 32) atomicAdd(&dst[c], src[c] * scale);
}

// ---------- timestep embedding: silu(emb) @ emb_w + emb_b -> [8,256] ----------
__global__ void k_emb(const float* __restrict__ emb, const float* __restrict__ W,
                      const float* __restrict__ bias, float* __restrict__ outp) {
    __shared__ float se[EMB_CH];
    const int b = blockIdx.x, o = threadIdx.x;
    for (int k = o; k < EMB_CH; k += 256) se[k] = silu(emb[b * EMB_CH + k]);
    __syncthreads();
    float acc = bias[o];
    for (int k = 0; k < EMB_CH; ++k) acc += se[k] * W[k * C + o];
    outp[b * C + o] = acc;
}

// ---------- weights -> bf16, transposed + padded: WT[256][1856] ----------
__global__ void k_padWT(const float* __restrict__ W, __bf16* __restrict__ WT) {
    const int o = blockIdx.x;
    for (int kk = threadIdx.x; kk < KPP; kk += 256) {
        float v = 0.f;
        if (kk < KP) {
            int t = kk / CPAD, c = kk - t * CPAD;
            if (c < CP) v = W[(size_t)(t * CP + c) * C + o];
        }
        WT[(size_t)o * KPP + kk] = (__bf16)v;
    }
}

// ---------- WMMA GEMM with async double-buffered LDS pipeline ----------
// mode 0: out = gemm + embout[batch_id[n]] ; mode 1: out = gemm + x[n]
#define BM 64
#define BN 128
#define LDAF 40   // floats per A row in LDS
#define LDBH 40   // bf16 per B row in LDS

__device__ __forceinline__ void stage_tile(const float* __restrict__ Srow0,
                                           const __bf16* __restrict__ WT0,
                                           float* __restrict__ Abuf,
                                           __bf16* __restrict__ Bbuf,
                                           int tid, int kt) {
#pragma unroll
    for (int i = 0; i < 2; ++i) {             // A: 64 rows x 32 f32, 16B per lane
        int e = tid + i * 256;
        int r = e >> 3, q = e & 7;
        const float* g = Srow0 + (size_t)r * KPP + kt * 32 + q * 4;
        float* l = Abuf + r * LDAF + q * 4;
#if ASYNC_OK
        __builtin_amdgcn_global_load_async_to_lds_b128(AS1V(g), AS3V(l), 0, 0);
#else
        *(float4*)l = *(const float4*)g;
#endif
    }
#pragma unroll
    for (int i = 0; i < 2; ++i) {             // B: 128 cols x 32 bf16, 16B per lane
        int e = tid + i * 256;
        int o = e >> 2, q = e & 3;
        const __bf16* g = WT0 + (size_t)o * KPP + kt * 32 + q * 8;
        __bf16* l = Bbuf + o * LDBH + q * 8;
#if ASYNC_OK
        __builtin_amdgcn_global_load_async_to_lds_b128(AS1V(g), AS3V(l), 0, 0);
#else
        *(U4*)l = *(const U4*)g;
#endif
    }
}

__device__ __forceinline__ void wait_async_lds() {
#if ASYNC_OK
    __builtin_amdgcn_s_wait_asynccnt(0);
#endif
}

__global__ __launch_bounds__(256) void k_gemm(
        const float* __restrict__ S, const __bf16* __restrict__ WT,
        const int* __restrict__ batch_id, const float* __restrict__ addvec,
        float* __restrict__ out, int mode) {
    __shared__ float  As[2][BM * LDAF];
    __shared__ __bf16 Bs[2][BN * LDBH];
    const int tid = threadIdx.x;
    const int n0  = blockIdx.x * BM;
    const int bn0 = blockIdx.y * BN;
    const int wave = tid >> 5, lane = tid & 31;
    const int wr = wave & 3, wc = wave >> 2;
    const int M16 = lane & 15, hl = lane >> 4;

    const float*  Srow0 = S + (size_t)n0 * KPP;
    const __bf16* WT0   = WT + (size_t)bn0 * KPP;

    v8f acc[4];
#pragma unroll
    for (int j = 0; j < 4; ++j)
#pragma unroll
        for (int r = 0; r < 8; ++r) acc[j][r] = 0.f;

    stage_tile(Srow0, WT0, As[0], Bs[0], tid, 0);

    for (int kt = 0; kt < NKT; ++kt) {
        wait_async_lds();
        __syncthreads();
        if (kt + 1 < NKT)
            stage_tile(Srow0, WT0, As[(kt + 1) & 1], Bs[(kt + 1) & 1], tid, kt + 1);

        const float*  Ab = As[kt & 1];
        const __bf16* Bb = Bs[kt & 1];

        // A fragment: row = wr*16+M16 ; K = {8h..8h+7, 16+8h..16+8h+7} ; cvt f32->bf16
        const float* ap = Ab + (wr * 16 + M16) * LDAF;
        float4 f0 = *(const float4*)(ap + 8 * hl);
        float4 f1 = *(const float4*)(ap + 8 * hl + 4);
        float4 f2 = *(const float4*)(ap + 16 + 8 * hl);
        float4 f3 = *(const float4*)(ap + 20 + 8 * hl);
        FragBF a;
        a.v = (v16bf){(__bf16)f0.x, (__bf16)f0.y, (__bf16)f0.z, (__bf16)f0.w,
                      (__bf16)f1.x, (__bf16)f1.y, (__bf16)f1.z, (__bf16)f1.w,
                      (__bf16)f2.x, (__bf16)f2.y, (__bf16)f2.z, (__bf16)f2.w,
                      (__bf16)f3.x, (__bf16)f3.y, (__bf16)f3.z, (__bf16)f3.w};
#pragma unroll
        for (int j = 0; j < 4; ++j) {
            FragBF b;
            const __bf16* bp = Bb + (wc * 64 + j * 16 + M16) * LDBH + 8 * hl;
            b.u[0] = *(const U4*)bp;
            b.u[1] = *(const U4*)(bp + 16);
            acc[j] = __builtin_amdgcn_wmma_f32_16x16x32_bf16(
                false, a.v, false, b.v, (short)0, acc[j], false, false);
        }
    }

    // epilogue: D lane layout M = r + 8*hl, Ncol = M16
#pragma unroll
    for (int j = 0; j < 4; ++j) {
#pragma unroll
        for (int r = 0; r < 8; ++r) {
            int node = n0 + wr * 16 + r + 8 * hl;
            int colo = bn0 + wc * 64 + j * 16 + M16;
            if (node < N_NODES) {
                float v = acc[j][r];
                if (mode == 0) v += addvec[batch_id[node] * C + colo];
                else           v += addvec[(size_t)node * C + colo];
                out[(size_t)node * C + colo] = v;
            }
        }
    }
}

// ================= host side =================
extern "C" void kernel_launch(void* const* d_in, const int* in_sizes, int n_in,
                              void* d_out, int out_size, void* d_ws, size_t ws_size,
                              hipStream_t stream) {
    const float* x        = (const float*)d_in[0];
    const float* emb      = (const float*)d_in[1];
    const int*   ei       = (const int*)d_in[2];
    const int*   ei_row   = ei;
    const int*   ei_col   = ei + N_EDGES;
    const int*   etype    = (const int*)d_in[3];
    const int*   ntype    = (const int*)d_in[4];
    const int*   batch_id = (const int*)d_in[5];
    const float* gn1_w    = (const float*)d_in[6];
    const float* gn1_b    = (const float*)d_in[7];
    const float* conv1_w  = (const float*)d_in[8];
    const float* emb_w    = (const float*)d_in[9];
    const float* emb_b    = (const float*)d_in[10];
    const float* gn2_w    = (const float*)d_in[11];
    const float* gn2_b    = (const float*)d_in[12];
    const float* conv2_w  = (const float*)d_in[13];
    float* out            = (float*)d_out;

    char* ws = (char*)d_ws;
    size_t off = 0;
    auto alloc = [&](size_t bytes) { void* p = ws + off; off += (bytes + 255) & ~(size_t)255; return p; };

    float* ssum1  = (float*)alloc(256 * 4);
    float* ssq1   = (float*)alloc(256 * 4);
    float* mean1  = (float*)alloc(256 * 4);
    float* istd1  = (float*)alloc(256 * 4);
    float* ssum2  = (float*)alloc(256 * 4);
    float* ssq2   = (float*)alloc(256 * 4);
    float* mean2  = (float*)alloc(256 * 4);
    float* istd2  = (float*)alloc(256 * 4);
    int*   bcnt   = (int*)  alloc(BATCH * 4);
    float* embout = (float*)alloc(BATCH * C * 4);
    float* Sbuf   = (float*)alloc((size_t)NPAD * KPP * 4);        // 742.6 MB
    float* h1     = (float*)alloc((size_t)N_NODES * CPAD * 4);    // 105.6 MB
    float* hconv  = (float*)alloc((size_t)N_NODES * C * 4);       // 102.4 MB
    int*   cnt    = (int*)  alloc((size_t)N_NODES * NET * 4);
    float* invc   = (float*)alloc((size_t)N_NODES * NET * 4);
    __bf16* WT1   = (__bf16*)alloc((size_t)C * KPP * 2);
    __bf16* WT2   = (__bf16*)alloc((size_t)C * KPP * 2);

    (void)hipMemsetAsync(ssum1, 0, 256 * 4, stream);
    (void)hipMemsetAsync(ssq1, 0, 256 * 4, stream);
    (void)hipMemsetAsync(ssum2, 0, 256 * 4, stream);
    (void)hipMemsetAsync(ssq2, 0, 256 * 4, stream);
    (void)hipMemsetAsync(bcnt, 0, BATCH * 4, stream);
    (void)hipMemsetAsync(cnt, 0, (size_t)N_NODES * NET * 4, stream);
    (void)hipMemsetAsync(Sbuf, 0, (size_t)NPAD * KPP * 4, stream);

    const int TB = 256;
    // GN1
    k_batch_count<<<64, TB, 0, stream>>>(batch_id, bcnt);
    k_stats<<<(N_NODES + 127) / 128, TB, 0, stream>>>(x, batch_id, ssum1, ssq1);
    k_finalize<<<1, TB, 0, stream>>>(ssum1, ssq1, bcnt, mean1, istd1);
    k_norm_act<<<N_NODES, TB, 0, stream>>>(x, batch_id, ntype, mean1, istd1, gn1_w, gn1_b, h1);
    // edge counts / inverse (shared by both convs)
    k_count_edges<<<(N_EDGES + TB - 1) / TB, TB, 0, stream>>>(ei_row, etype, cnt);
    k_invcnt<<<(N_NODES * NET + TB - 1) / TB, TB, 0, stream>>>(cnt, invc);
    // conv1 scatter (pre-scaled)
    k_scatter<<<(N_EDGES * 32 + TB - 1) / TB, TB, 0, stream>>>(h1, ei_row, ei_col, etype, invc, Sbuf);
    // embedding + weight prep
    k_emb<<<BATCH, TB, 0, stream>>>(emb, emb_w, emb_b, embout);
    k_padWT<<<C, TB, 0, stream>>>(conv1_w, WT1);
    k_padWT<<<C, TB, 0, stream>>>(conv2_w, WT2);
    // GEMM1 (+ emb add) -> hconv
    dim3 ggrid((N_NODES + BM - 1) / BM, C / BN);
    k_gemm<<<ggrid, TB, 0, stream>>>(Sbuf, WT1, batch_id, embout, hconv, 0);
    // GN2
    k_stats<<<(N_NODES + 127) / 128, TB, 0, stream>>>(hconv, batch_id, ssum2, ssq2);
    k_finalize<<<1, TB, 0, stream>>>(ssum2, ssq2, bcnt, mean2, istd2);
    k_norm_act<<<N_NODES, TB, 0, stream>>>(hconv, batch_id, ntype, mean2, istd2, gn2_w, gn2_b, h1);
    // conv2 scatter (re-zero S)
    (void)hipMemsetAsync(Sbuf, 0, (size_t)NPAD * KPP * 4, stream);
    k_scatter<<<(N_EDGES * 32 + TB - 1) / TB, TB, 0, stream>>>(h1, ei_row, ei_col, etype, invc, Sbuf);
    // GEMM2 (+ residual x) -> out
    k_gemm<<<ggrid, TB, 0, stream>>>(Sbuf, WT2, batch_id, x, out, 1);

    (void)in_sizes; (void)n_in; (void)out_size; (void)ws_size;
}
